// MHSelfAttention_27539330302473
// MI455X (gfx1250) — compile-verified, matrix operand order checked by reading, not run on hardware
//
#include <hip/hip_runtime.h>
#include <hip/hip_bf16.h>

typedef __attribute__((ext_vector_type(16))) _Float16 v16h;
typedef __attribute__((ext_vector_type(8)))  float    v8f;

union Frag {
    v16h h;
    uint4 u[2];
};

__device__ __forceinline__ v8f wmma_f16(v16h a, v16h b, v8f c) {
    // D = A(16x32 f16) * B(32x16 f16) + C(16x16 f32)
    return __builtin_amdgcn_wmma_f32_16x16x32_f16(
        /*neg_a=*/false, a, /*neg_b=*/false, b,
        /*c_mod=*/(short)0, c, /*reuse_a=*/false, /*reuse_b=*/false);
}

// ---------------------------------------------------------------------------
// Problem constants
// ---------------------------------------------------------------------------
#define BB 2
#define SS 2048
#define DD 1024
#define HH 16
#define DHH 64
#define MTOT (BB * SS)          // 4096 rows through both GEMMs
#define N_QKV (3 * DD)          // 3072
#define INV_SQRT_DH 0.125f      // 1/sqrt(64)

// ---------------------------------------------------------------------------
// Convert f32 -> f16 (for x / attn inputs)
// ---------------------------------------------------------------------------
__global__ __launch_bounds__(256) void cvt_f32_to_f16(const float* __restrict__ in,
                                                      _Float16* __restrict__ out, int n) {
    int i = blockIdx.x * 256 + threadIdx.x;
    if (i < n) out[i] = (_Float16)in[i];
}

// Transpose+convert: in[R][C] f32 -> out[C][R] f16, LDS-tiled 32x32.
__global__ __launch_bounds__(256) void transpose_cvt(const float* __restrict__ in,
                                                     _Float16* __restrict__ out,
                                                     int R, int C) {
    __shared__ _Float16 tile[32][33];
    const int c0 = blockIdx.x * 32, r0 = blockIdx.y * 32;
    const int tx = threadIdx.x, ty = threadIdx.y;   // 32 x 8
#pragma unroll
    for (int i = 0; i < 4; ++i)
        tile[ty + i * 8][tx] = (_Float16)in[(size_t)(r0 + ty + i * 8) * C + c0 + tx];
    __syncthreads();
#pragma unroll
    for (int i = 0; i < 4; ++i)
        out[(size_t)(c0 + ty + i * 8) * R + r0 + tx] = tile[tx][ty + i * 8];
}

// ---------------------------------------------------------------------------
// Shared WMMA GEMM mainloop: C[128x128] tile = A[m0:,:] * Bt[n0:,:]^T
// A row-major [M][K] f16, Bt row-major [N][K] f16 (i.e. B pre-transposed).
// 256 threads = 8 waves in a 4(M) x 2(N) grid; each wave: 32x64 (2x4 WMMA tiles).
// ---------------------------------------------------------------------------
__device__ __forceinline__ void gemm_mainloop(const _Float16* __restrict__ A,
                                              const _Float16* __restrict__ Bt,
                                              int K, int m0, int n0,
                                              _Float16* ldsA, _Float16* ldsB,
                                              v8f (&acc)[2][4]) {
    const int tid  = threadIdx.x;
    const int lane = tid & 31;
    const int wave = tid >> 5;
    const int l16  = lane & 15;
    const int hi   = lane >> 4;       // 0: lanes 0-15, 1: lanes 16-31
    const int wm   = wave >> 1;       // 0..3 -> 32-row slabs
    const int wn   = wave & 1;        // 0..1 -> 64-col slabs

    for (int k0 = 0; k0 < K; k0 += 32) {
        // Cooperative tile load: 128x32 halves each = 512 uint4 per matrix.
#pragma unroll
        for (int it = 0; it < 2; ++it) {
            const int c   = tid + 256 * it;
            const int row = c >> 2;
            const int seg = c & 3;
            ((uint4*)ldsA)[c] = *(const uint4*)(A  + (size_t)(m0 + row) * K + k0 + seg * 8);
            ((uint4*)ldsB)[c] = *(const uint4*)(Bt + (size_t)(n0 + row) * K + k0 + seg * 8);
        }
        if (k0 + 32 < K) {
            // speculative prefetch of the next K-slab (global_prefetch_b8)
            __builtin_prefetch(A  + (size_t)(m0 + (tid >> 1)) * K + k0 + 32, 0, 3);
            __builtin_prefetch(Bt + (size_t)(n0 + (tid >> 1)) * K + k0 + 32, 0, 3);
        }
        __syncthreads();

        Frag fa[2], fb[4];
#pragma unroll
        for (int mt = 0; mt < 2; ++mt) {
            const uint4* p = (const uint4*)(ldsA + (wm * 32 + mt * 16 + l16) * 32);
            fa[mt].u[0] = p[hi];
            fa[mt].u[1] = p[2 + hi];
        }
#pragma unroll
        for (int nt = 0; nt < 4; ++nt) {
            const uint4* p = (const uint4*)(ldsB + (wn * 64 + nt * 16 + l16) * 32);
            fb[nt].u[0] = p[hi];
            fb[nt].u[1] = p[2 + hi];
        }
#pragma unroll
        for (int mt = 0; mt < 2; ++mt)
#pragma unroll
            for (int nt = 0; nt < 4; ++nt)
                acc[mt][nt] = wmma_f16(fa[mt].h, fb[nt].h, acc[mt][nt]);
        __syncthreads();
    }
}

// ---------------------------------------------------------------------------
// GEMM 1: qkv = xh @ W_qkv + b_qkv, scattered into Q, K ([BH][S][64]) and
// V transposed ([BH][64][S]); scale 1/sqrt(dh) folded into Q.
// Grid: (MTOT/128, N_QKV/128) = (32, 24)
// ---------------------------------------------------------------------------
__global__ __launch_bounds__(256) void gemm_qkv_kernel(const _Float16* __restrict__ xh,
                                                       const _Float16* __restrict__ WqkvT,
                                                       const float* __restrict__ bias,
                                                       _Float16* __restrict__ Qb,
                                                       _Float16* __restrict__ Kb,
                                                       _Float16* __restrict__ Vt) {
    __shared__ _Float16 lds[2 * 128 * 32];
    v8f acc[2][4] = {};
    const int m0 = blockIdx.x * 128;
    const int n0 = blockIdx.y * 128;
    gemm_mainloop(xh, WqkvT, DD, m0, n0, lds, lds + 4096, acc);

    const int tid = threadIdx.x, lane = tid & 31, wave = tid >> 5;
    const int l16 = lane & 15, hi = lane >> 4;
    const int wm = wave >> 1, wn = wave & 1;

#pragma unroll
    for (int nt = 0; nt < 4; ++nt) {
        const int col   = n0 + wn * 64 + nt * 16 + l16;   // 0..3071
        const float bv  = bias[col];
        const int which = col >> 10;       // 0=Q 1=K 2=V
        const int d     = col & (DD - 1);
        const int hh    = d >> 6;
        const int dh    = d & 63;
#pragma unroll
        for (int mt = 0; mt < 2; ++mt) {
#pragma unroll
            for (int r = 0; r < 8; ++r) {
                const int row = m0 + wm * 32 + mt * 16 + r + hi * 8;
                const int b   = row >> 11;           // /S
                const int s   = row & (SS - 1);
                const float v = acc[mt][nt][r] + bv;
                const size_t bh = (size_t)(b * HH + hh);
                if (which == 0)
                    Qb[(bh * SS + s) * DHH + dh] = (_Float16)(v * INV_SQRT_DH);
                else if (which == 1)
                    Kb[(bh * SS + s) * DHH + dh] = (_Float16)v;
                else
                    Vt[(bh * DHH + dh) * SS + s] = (_Float16)v;
            }
        }
    }
}

// ---------------------------------------------------------------------------
// Flash attention: grid = BH * (S/64) blocks, 128 threads (4 waves).
// Each wave owns 16 query rows; streams 32 causal keys per iteration:
//   4x wmma for scores (dh=64 -> 2 K=32 steps x 2 key-column tiles),
//   online softmax with shfl_xor row reductions,
//   P staged through LDS (C-layout -> A-layout), 4x wmma for P*V.
// ---------------------------------------------------------------------------
__global__ __launch_bounds__(128) void attn_kernel(const _Float16* __restrict__ Qb,
                                                   const _Float16* __restrict__ Kb,
                                                   const _Float16* __restrict__ Vt,
                                                   _Float16* __restrict__ attn_h) {
    __shared__ _Float16 ldsP[4 * 16 * 32];
    const int tid  = threadIdx.x;
    const int lane = tid & 31;
    const int wave = tid >> 5;
    const int l16  = lane & 15;
    const int hi   = lane >> 4;
    const int bh   = blockIdx.x >> 5;        // 0..31  (b*16+h)
    const int qg   = blockIdx.x & 31;
    const int q0   = qg * 64 + wave * 16;    // this wave's query-row base
    const int b    = bh >> 4;
    const int h    = bh & 15;
    _Float16* P = ldsP + wave * 512;

    // Q fragments (A-layout), dh 0..31 and 32..63
    Frag aq0, aq1;
    {
        const uint4* pq = (const uint4*)(Qb + ((size_t)bh * SS + q0 + l16) * DHH);
        aq0.u[0] = pq[hi];     aq0.u[1] = pq[2 + hi];
        aq1.u[0] = pq[4 + hi]; aq1.u[1] = pq[6 + hi];
    }

    v8f acc[4] = {};             // 16 rows x 64 dh
    float m_r[8], l_r[8];
#pragma unroll
    for (int r = 0; r < 8; ++r) { m_r[r] = -1.0e30f; l_r[r] = 0.0f; }

    const int ntiles = (q0 + 47) >> 5;       // causal: keys <= q0+15
    for (int j = 0; j < ntiles; ++j) {
        const int kb = j << 5;

        // ---- scores: two 16x16 tiles over keys [kb, kb+32) ----
        v8f st[2];
#pragma unroll
        for (int t = 0; t < 2; ++t) {
            Frag kf0, kf1;
            const uint4* pk = (const uint4*)(Kb + ((size_t)bh * SS + kb + t * 16 + l16) * DHH);
            kf0.u[0] = pk[hi];     kf0.u[1] = pk[2 + hi];
            kf1.u[0] = pk[4 + hi]; kf1.u[1] = pk[6 + hi];
            v8f z = {};
            z = wmma_f16(aq0.h, kf0.h, z);
            z = wmma_f16(aq1.h, kf1.h, z);
            st[t] = z;
        }

        // ---- causal mask + online softmax ----
        const int key0 = kb + l16;
        const int key1 = kb + 16 + l16;
#pragma unroll
        for (int r = 0; r < 8; ++r) {
            const int qrow = q0 + r + hi * 8;
            float s0 = st[0][r]; if (key0 > qrow) s0 = -1.0e4f;
            float s1 = st[1][r]; if (key1 > qrow) s1 = -1.0e4f;
            float mx = fmaxf(s0, s1);
            mx = fmaxf(mx, __shfl_xor(mx, 1, 32));
            mx = fmaxf(mx, __shfl_xor(mx, 2, 32));
            mx = fmaxf(mx, __shfl_xor(mx, 4, 32));
            mx = fmaxf(mx, __shfl_xor(mx, 8, 32));
            const float mnew = fmaxf(m_r[r], mx);
            const float p0 = __expf(s0 - mnew);
            const float p1 = __expf(s1 - mnew);
            const float corr = __expf(m_r[r] - mnew);
            float rs = p0 + p1;
            rs += __shfl_xor(rs, 1, 32);
            rs += __shfl_xor(rs, 2, 32);
            rs += __shfl_xor(rs, 4, 32);
            rs += __shfl_xor(rs, 8, 32);
            l_r[r] = l_r[r] * corr + rs;
            m_r[r] = mnew;
#pragma unroll
            for (int c = 0; c < 4; ++c) acc[c][r] = acc[c][r] * corr;
            // stage probabilities: row-major 16x32 f16 per wave
            const int prow = r + hi * 8;
            P[prow * 32 + l16]      = (_Float16)p0;
            P[prow * 32 + 16 + l16] = (_Float16)p1;
        }
        __builtin_amdgcn_wave_barrier();
        asm volatile("s_wait_dscnt 0x0" ::: "memory");

        // ---- P (A-layout) from LDS, P*V into acc ----
        Frag fp;
        {
            const uint4* pp = (const uint4*)(P + l16 * 32);
            fp.u[0] = pp[hi];
            fp.u[1] = pp[2 + hi];
        }
#pragma unroll
        for (int c = 0; c < 4; ++c) {
            Frag fv;
            const uint4* pv =
                (const uint4*)(Vt + ((size_t)bh * DHH + c * 16 + l16) * SS + kb);
            fv.u[0] = pv[hi];
            fv.u[1] = pv[2 + hi];
            acc[c] = wmma_f16(fp.h, fv.h, acc[c]);
        }
        __builtin_amdgcn_wave_barrier();
    }

    // ---- normalize and scatter to attn_h [B*S][D] f16 ----
#pragma unroll
    for (int r = 0; r < 8; ++r) {
        const float rl = 1.0f / l_r[r];
        const int qrow = q0 + r + hi * 8;
#pragma unroll
        for (int c = 0; c < 4; ++c) {
            const int dh = c * 16 + l16;
            attn_h[((size_t)(b * SS + qrow)) * DD + h * DHH + dh] =
                (_Float16)(acc[c][r] * rl);
        }
    }
}

// ---------------------------------------------------------------------------
// GEMM 2: out = attn_h @ W_out + b_out (f32 output). Grid (32, 8).
// ---------------------------------------------------------------------------
__global__ __launch_bounds__(256) void gemm_out_kernel(const _Float16* __restrict__ Ah,
                                                       const _Float16* __restrict__ WoutT,
                                                       const float* __restrict__ bias,
                                                       float* __restrict__ out) {
    __shared__ _Float16 lds[2 * 128 * 32];
    v8f acc[2][4] = {};
    const int m0 = blockIdx.x * 128;
    const int n0 = blockIdx.y * 128;
    gemm_mainloop(Ah, WoutT, DD, m0, n0, lds, lds + 4096, acc);

    const int tid = threadIdx.x, lane = tid & 31, wave = tid >> 5;
    const int l16 = lane & 15, hi = lane >> 4;
    const int wm = wave >> 1, wn = wave & 1;

#pragma unroll
    for (int nt = 0; nt < 4; ++nt) {
        const int col  = n0 + wn * 64 + nt * 16 + l16;
        const float bv = bias[col];
#pragma unroll
        for (int mt = 0; mt < 2; ++mt) {
#pragma unroll
            for (int r = 0; r < 8; ++r) {
                const int row = m0 + wm * 32 + mt * 16 + r + hi * 8;
                out[(size_t)row * DD + col] = acc[mt][nt][r] + bv;
            }
        }
    }
}

// ---------------------------------------------------------------------------
// Launch
// ---------------------------------------------------------------------------
extern "C" void kernel_launch(void* const* d_in, const int* in_sizes, int n_in,
                              void* d_out, int out_size, void* d_ws, size_t ws_size,
                              hipStream_t stream) {
    const float* x      = (const float*)d_in[0];   // [B,S,D]
    const float* W_qkv  = (const float*)d_in[1];   // [D, 3D]
    const float* b_qkv  = (const float*)d_in[2];   // [3D]
    const float* W_out  = (const float*)d_in[3];   // [D, D]
    const float* b_out  = (const float*)d_in[4];   // [D]
    float* out = (float*)d_out;                    // [B,S,D] f32

    char* ws = (char*)d_ws;
    const size_t MB = 1024 * 1024;
    _Float16* xh    = (_Float16*)(ws);                 //  8 MB: x f16 [4096][1024]
    _Float16* WqkvT = (_Float16*)(ws + 8 * MB);        //  6 MB: W_qkv^T f16 [3072][1024]
    _Float16* WoutT = (_Float16*)(ws + 14 * MB);       //  2 MB: W_out^T f16 [1024][1024]
    _Float16* Qb    = (_Float16*)(ws + 16 * MB);       //  8 MB: Q f16 [BH][S][64] (pre-scaled)
    _Float16* Kb    = (_Float16*)(ws + 24 * MB);       //  8 MB: K f16 [BH][S][64]
    _Float16* Vt    = (_Float16*)(ws + 32 * MB);       //  8 MB: V^T f16 [BH][64][S]
    _Float16* Ah    = (_Float16*)(ws + 40 * MB);       //  8 MB: attn f16 [4096][1024]

    // 1) precision conversion / weight transposes
    cvt_f32_to_f16<<<(MTOT * DD) / 256, 256, 0, stream>>>(x, xh, MTOT * DD);
    transpose_cvt<<<dim3(N_QKV / 32, DD / 32), dim3(32, 8), 0, stream>>>(W_qkv, WqkvT, DD, N_QKV);
    transpose_cvt<<<dim3(DD / 32, DD / 32), dim3(32, 8), 0, stream>>>(W_out, WoutT, DD, DD);

    // 2) QKV projection (WMMA GEMM) with scatter into Q / K / V^T
    gemm_qkv_kernel<<<dim3(MTOT / 128, N_QKV / 128), 256, 0, stream>>>(
        xh, WqkvT, b_qkv, Qb, Kb, Vt);

    // 3) causal flash attention (WMMA QK^T and PV)
    attn_kernel<<<BB * HH * (SS / 64), 128, 0, stream>>>(Qb, Kb, Vt, Ah);

    // 4) output projection (WMMA GEMM), f32 epilogue with bias
    gemm_out_kernel<<<dim3(MTOT / 128, DD / 128), 256, 0, stream>>>(Ah, WoutT, b_out, out);
}
